// CrossHeadAttention_82858509074851
// MI455X (gfx1250) — matrix operand order repaired and checked by reading, not compiled, not run on hardware
//
#include <hip/hip_runtime.h>

typedef __attribute__((ext_vector_type(16))) __bf16 v16bf;
typedef __attribute__((ext_vector_type(8)))  __bf16 v8bf;
typedef __attribute__((ext_vector_type(8)))  float  v8f;
typedef __attribute__((ext_vector_type(4))) unsigned int u32x4;
typedef __attribute__((ext_vector_type(8))) int i32x8;
typedef __attribute__((ext_vector_type(4))) int i32x4;

#define N_EMBD 1024
#define NUM_HEADS 16
#define HEAD_SIZE 64
#define BATCH 2
#define SEQ_T 2048
#define SEQ_S 2048

__device__ __forceinline__ __bf16 f2bf(float f) { return (__bf16)f; }

// ---------------------------------------------------------------------------
// f32 -> bf16 conversion (grid-stride)
// ---------------------------------------------------------------------------
__global__ __launch_bounds__(256) void cvt_f32_bf16(const float* __restrict__ in,
                                                    __bf16* __restrict__ out, int n) {
  int i = blockIdx.x * blockDim.x + threadIdx.x;
  int stride = gridDim.x * blockDim.x;
  for (; i < n; i += stride) out[i] = f2bf(in[i]);
}

// ---------------------------------------------------------------------------
// bf16 GEMM: C[M,1024] = A[M,1024] @ B[1024,1024], all row-major.
// 256 threads = 8 waves. Block tile 128x128, BK=32.
// A-tile staged via Tensor Data Mover (tensor_load_to_lds + s_wait_tensorcnt);
// B-tile staged transposed by VALU/ds (TDM cannot transpose).
// Wave grid 4(M) x 2(N): each wave computes 32x64 = 2x4 WMMA tiles.
// ---------------------------------------------------------------------------
__global__ __launch_bounds__(256) void gemm_bf16_kernel(const __bf16* __restrict__ A,
                                                        const __bf16* __restrict__ B,
                                                        __bf16* __restrict__ C, int M) {
  const int K = 1024, N = 1024;
  __shared__ __align__(16) __bf16 As[128][32];   // [m][k] -- TDM tile, no pad
  __shared__ __align__(16) __bf16 Bs[128][40];   // [n][k] (transposed), padded

  int tid  = threadIdx.x;
  int lane = tid & 31;
  int wave = tid >> 5;
  int lm   = lane & 15;
  int lh   = lane >> 4;
  int wy   = wave & 3;    // 4 waves along M, 32 rows each
  int wx   = wave >> 2;   // 2 waves along N, 64 cols each
  int bm   = blockIdx.x * 128;
  int bn   = blockIdx.y * 128;

  int bkr  = tid >> 3, bnc = (tid & 7) * 16;       // B fill: 32 rows x 128 cols

  // Constant parts of the Tensor DMA descriptor (D#) for the A tile:
  // 2D tile: tile_dim0=32 (k, contiguous), tile_dim1=128 (rows),
  // tensor_dim0=1024, tensor_dim1=M, tensor_dim0_stride=1024, data_size=2B.
  unsigned int ldsAddr = (unsigned int)(size_t)(&As[0][0]);
  i32x8 g1;
  g1[0] = (int)(1u << 16);                 // data_size=1 (2 bytes); mask/flags=0
  g1[1] = (int)(1024u << 16);              // tensor_dim0[15:0] -> bits 63:48
  g1[2] = (int)(((unsigned)M & 0xFFFFu) << 16);  // tensor_dim1[15:0] -> bits 95:80
  g1[3] = (int)(32u << 16);                // tensor_dim0[31:16]=0 | tile_dim0=32
  g1[4] = 128;                             // tile_dim1=128, tile_dim2=0
  g1[5] = 1024;                            // tensor_dim0_stride[31:0]
  g1[6] = 0;                               // dim0_stride[47:32]=0, dim1_stride lo=0
  g1[7] = 0;
  i32x4 g2 = {0, 0, 0, 0};
  i32x4 g3 = {0, 0, 0, 0};
  i32x8 g4 = {0, 0, 0, 0, 0, 0, 0, 0};

  v8f acc[2][4] = {};

  for (int k0 = 0; k0 < K; k0 += 32) {
    // --- stage A tile via TDM (wave 0 issues; EXEC-independent DMA) ---
    if (wave == 0) {
      unsigned long long gaddr =
          (unsigned long long)(size_t)(A + (size_t)bm * K + k0);
      u32x4 g0;
      g0[0] = 1u;                                  // count=1, is_restore=0
      g0[1] = ldsAddr;                             // lds_addr (bits 63:32)
      g0[2] = (unsigned int)(gaddr & 0xFFFFFFFFu); // global_addr lo
      g0[3] = (unsigned int)((gaddr >> 32) & 0x01FFFFFFu) | (2u << 30); // type=2
      __builtin_amdgcn_tensor_load_to_lds(g0, g1, g2, g3, g4, 0);
    }
    // --- stage B tile, transposed into Bs[n][k] (coalesced global reads) ---
    const __bf16* bg = B + (size_t)(k0 + bkr) * N + bn + bnc;
    v8bf b0 = *(const v8bf*)(bg);
    v8bf b1 = *(const v8bf*)(bg + 8);
    if (k0 + 32 < K) __builtin_prefetch(bg + 32 * (size_t)N, 0, 1);
#pragma unroll
    for (int j = 0; j < 8; ++j) {
      Bs[bnc + j][bkr]     = b0[j];
      Bs[bnc + 8 + j][bkr] = b1[j];
    }
    if (wave == 0) __builtin_amdgcn_s_wait_tensorcnt(0);
    __syncthreads();

    // --- fragments ---
    v16bf af[2], bfg[4];
#pragma unroll
    for (int mt = 0; mt < 2; ++mt) {
      int r = wy * 32 + mt * 16 + lm;
      v8bf lo = *(const v8bf*)&As[r][lh * 8];
      v8bf hi = *(const v8bf*)&As[r][lh * 8 + 16];
#pragma unroll
      for (int j = 0; j < 8; ++j) { af[mt][j] = lo[j]; af[mt][j + 8] = hi[j]; }
    }
#pragma unroll
    for (int nt = 0; nt < 4; ++nt) {
      int c = wx * 64 + nt * 16 + lm;
      v8bf lo = *(const v8bf*)&Bs[c][lh * 8];
      v8bf hi = *(const v8bf*)&Bs[c][lh * 8 + 16];
#pragma unroll
      for (int j = 0; j < 8; ++j) { bfg[nt][j] = lo[j]; bfg[nt][j + 8] = hi[j]; }
    }
#pragma unroll
    for (int mt = 0; mt < 2; ++mt)
#pragma unroll
      for (int nt = 0; nt < 4; ++nt)
        acc[mt][nt] = __builtin_amdgcn_wmma_f32_16x16x32_bf16(
            false, af[mt], false, bfg[nt], (short)0, acc[mt][nt], false, false);
    __syncthreads();
  }

  // --- writeback: C/D layout: lane l, vgpr r -> (M=(l>>4)*8+r, N=l&15) ---
#pragma unroll
  for (int mt = 0; mt < 2; ++mt)
#pragma unroll
    for (int nt = 0; nt < 4; ++nt)
#pragma unroll
      for (int r = 0; r < 8; ++r) {
        int row = bm + wy * 32 + mt * 16 + lh * 8 + r;
        int col = bn + wx * 64 + nt * 16 + lm;
        C[(size_t)row * N + col] = f2bf(acc[mt][nt][r]);
      }
}

// ---------------------------------------------------------------------------
// Fused attention: one block per (b, h, 16-query tile). 128 threads = 4 waves.
// Full 16x2048 f32 score slab lives in LDS (128KB; CDNA5 WGP has 320KB).
// Phase 1: scores = scale * Q Kt (WMMA, waves split S into 512-wide stripes)
// Phase 2: block-wide softmax over LDS (unnormalized exp + 1/sum)
// Phase 3: out = P V (WMMA per wave stripe), cross-wave reduce via LDS.
// ---------------------------------------------------------------------------
__global__ __launch_bounds__(128) void attn_kernel(const __bf16* __restrict__ Q,
                                                   const __bf16* __restrict__ K,
                                                   const __bf16* __restrict__ V,
                                                   float* __restrict__ out) {
  extern __shared__ char smemRaw[];
  float* scores  = (float*)smemRaw;            // 16 * 2048
  float* red     = scores + 16 * 2048;         // 16 * 8
  float* rmax    = red + 128;                  // 16
  float* rsumInv = rmax + 16;                  // 16
  float* partial = scores;                     // reused after phase 3: [4][16][64]

  const int HD = N_EMBD;
  int idx = blockIdx.x;
  int qt  = idx & 127;  idx >>= 7;   // SEQ_T/16 = 128 query tiles
  int h   = idx & 15;   idx >>= 4;
  int b   = idx;

  int tid  = threadIdx.x;
  int lane = tid & 31;
  int wave = tid >> 5;
  int lm   = lane & 15;
  int lh   = lane >> 4;
  const float scale = 0.03125f;  // 1024^-0.5

  // --- Q A-fragments for d=0..63 (two 16x32 slices), loaded once ---
  const __bf16* qbase = Q + ((size_t)(b * SEQ_T + qt * 16 + lm)) * HD + h * HEAD_SIZE;
  v16bf aq[2];
#pragma unroll
  for (int f = 0; f < 2; ++f) {
    v8bf lo = *(const v8bf*)(qbase + f * 32 + lh * 8);
    v8bf hi = *(const v8bf*)(qbase + f * 32 + 16 + lh * 8);
#pragma unroll
    for (int j = 0; j < 8; ++j) { aq[f][j] = lo[j]; aq[f][j + 8] = hi[j]; }
  }

  // --- Phase 1: scores = scale * (Q K^T), wave w owns S stripe [w*512, w*512+512) ---
  for (int st = 0; st < 32; ++st) {
    int s0 = wave * 512 + st * 16;
    const __bf16* kb = K + ((size_t)(b * SEQ_S + s0 + lm)) * HD + h * HEAD_SIZE;
    v8f acc = {};
#pragma unroll
    for (int f = 0; f < 2; ++f) {
      v16bf bk;  // B frag: lane l = column s0+lm, d at f*32 + lh*8 and f*32 + 16 + lh*8
      v8bf lo = *(const v8bf*)(kb + f * 32 + lh * 8);
      v8bf hi = *(const v8bf*)(kb + f * 32 + 16 + lh * 8);
#pragma unroll
      for (int j = 0; j < 8; ++j) { bk[j] = lo[j]; bk[j + 8] = hi[j]; }
      acc = __builtin_amdgcn_wmma_f32_16x16x32_bf16(false, aq[f], false, bk,
                                                    (short)0, acc, false, false);
    }
#pragma unroll
    for (int r = 0; r < 8; ++r) {
      int m = lh * 8 + r;
      scores[m * 2048 + s0 + lm] = acc[r] * scale;
    }
  }
  __syncthreads();

  // --- Phase 2: softmax over S (rows partitioned 16 rows x 8 threads) ---
  {
    int row = tid >> 3;
    int sub = tid & 7;
    const float* srow = scores + row * 2048;
    float mx = -3.402823466e38f;
    for (int c = sub * 256; c < sub * 256 + 256; ++c) mx = fmaxf(mx, srow[c]);
    red[row * 8 + sub] = mx;
    __syncthreads();
    if (sub == 0) {
      float m2 = red[row * 8];
#pragma unroll
      for (int j = 1; j < 8; ++j) m2 = fmaxf(m2, red[row * 8 + j]);
      rmax[row] = m2;
    }
    __syncthreads();
    float rm = rmax[row];
    float* wrow = scores + row * 2048;
    float sum = 0.f;
    for (int c = sub * 256; c < sub * 256 + 256; ++c) {
      float e = __expf(wrow[c] - rm);
      wrow[c] = e;
      sum += e;
    }
    red[row * 8 + sub] = sum;
    __syncthreads();
    if (sub == 0) {
      float s2 = 0.f;
#pragma unroll
      for (int j = 0; j < 8; ++j) s2 += red[row * 8 + j];
      rsumInv[row] = 1.0f / s2;
    }
    __syncthreads();
  }

  // --- Phase 3: out = P V over this wave's S stripe ---
  v8f oacc[4] = {};
  for (int kc = 0; kc < 16; ++kc) {
    int sc = wave * 512 + kc * 32;
    v16bf ap;  // A frag from scores (f32 -> bf16): row lm, 16 contiguous k at lh*16
    {
      const float* sr = scores + lm * 2048 + sc;
      int kb = lh * 16;
#pragma unroll
      for (int j = 0; j < 8; ++j) {
        ap[j]     = f2bf(sr[kb + j]);
        ap[j + 8] = f2bf(sr[8 + kb + j]);
      }
    }
#pragma unroll
    for (int nt = 0; nt < 4; ++nt) {
      // B frag from V: column d = nt*16+lm, rows s = sc + lh*16 .. +15 (stride HD)
      v16bf bv;
      const __bf16* vb = V + ((size_t)(b * SEQ_S + sc + lh * 16)) * HD + h * HEAD_SIZE + nt * 16 + lm;
#pragma unroll
      for (int j = 0; j < 16; ++j) bv[j] = vb[(size_t)j * HD];
      oacc[nt] = __builtin_amdgcn_wmma_f32_16x16x32_bf16(false, ap, false, bv,
                                                         (short)0, oacc[nt], false, false);
    }
  }
  __syncthreads();  // all waves done reading scores; reuse slab for partials

#pragma unroll
  for (int nt = 0; nt < 4; ++nt)
#pragma unroll
    for (int r = 0; r < 8; ++r) {
      int m = lh * 8 + r;
      partial[(wave * 16 + m) * 64 + nt * 16 + lm] = oacc[nt][r];
    }
  __syncthreads();

  for (int i = tid; i < 16 * 64; i += 128) {
    int m = i >> 6, d = i & 63;
    float v = partial[m * 64 + d] + partial[(16 + m) * 64 + d] +
              partial[(32 + m) * 64 + d] + partial[(48 + m) * 64 + d];
    out[((size_t)(b * SEQ_T + qt * 16 + m)) * HD + h * HEAD_SIZE + d] = v * rsumInv[m];
  }
}

// ---------------------------------------------------------------------------
// Host launcher
// ---------------------------------------------------------------------------
extern "C" void kernel_launch(void* const* d_in, const int* in_sizes, int n_in,
                              void* d_out, int out_size, void* d_ws, size_t ws_size,
                              hipStream_t stream) {
  const float* x_enc = (const float*)d_in[0];
  const float* x     = (const float*)d_in[1];
  const float* Wk    = (const float*)d_in[2];
  const float* Wq    = (const float*)d_in[3];
  const float* Wv    = (const float*)d_in[4];
  float* out = (float*)d_out;

  const int M = BATCH * SEQ_T;            // 4096
  const int NM = M * N_EMBD;              // 4,194,304
  const int NW = N_EMBD * N_EMBD;         // 1,048,576

  char* ws = (char*)d_ws;
  size_t off = 0;
  auto carve = [&](size_t elems) -> __bf16* {
    __bf16* p = (__bf16*)(ws + off);
    off += ((elems * sizeof(__bf16) + 255) / 256) * 256;
    return p;
  };
  __bf16* xencBf = carve(NM);
  __bf16* xBf    = carve(NM);
  __bf16* wkBf   = carve(NW);
  __bf16* wqBf   = carve(NW);
  __bf16* wvBf   = carve(NW);
  __bf16* Kbf    = carve(NM);
  __bf16* Vbf    = carve(NM);
  __bf16* Qbf    = carve(NM);

  cvt_f32_bf16<<<4096, 256, 0, stream>>>(x_enc, xencBf, NM);
  cvt_f32_bf16<<<4096, 256, 0, stream>>>(x, xBf, NM);
  cvt_f32_bf16<<<2048, 256, 0, stream>>>(Wk, wkBf, NW);
  cvt_f32_bf16<<<2048, 256, 0, stream>>>(Wq, wqBf, NW);
  cvt_f32_bf16<<<2048, 256, 0, stream>>>(Wv, wvBf, NW);

  dim3 ggrid(M / 128, N_EMBD / 128);  // 32 x 8
  gemm_bf16_kernel<<<ggrid, 256, 0, stream>>>(xBf, wqBf, Qbf, M);
  gemm_bf16_kernel<<<ggrid, 256, 0, stream>>>(xencBf, wkBf, Kbf, M);
  gemm_bf16_kernel<<<ggrid, 256, 0, stream>>>(xencBf, wvBf, Vbf, M);

  size_t smem = (size_t)(16 * 2048 + 128 + 16 + 16) * sizeof(float);  // ~132KB
  int ablocks = BATCH * NUM_HEADS * (SEQ_T / 16);  // 4096
  attn_kernel<<<ablocks, 128, smem, stream>>>(Qbf, Kbf, Vbf, out);
}
